// QuantMamba2LMHeadModel_25477745999904
// MI455X (gfx1250) — matrix operand order repaired and checked by reading, not compiled, not run on hardware
//
#include <hip/hip_runtime.h>

#define D_MODEL 768
#define N_LAYER 8
#define D_STATE 128
#define D_CONV 4
#define HEADDIM 64
#define D_INNER 1536
#define NHEADS 24
#define D_IN_PROJ 3352   // 2*D_INNER + 2*D_STATE + NHEADS
#define CONV_DIM 1792    // D_INNER + 2*D_STATE
#define SEQ 1024
#define VOCAB 50288
#define EPS 1e-5f

typedef __attribute__((ext_vector_type(16))) __bf16 v16bf;
typedef __attribute__((ext_vector_type(8)))  float  v8f;
typedef __attribute__((ext_vector_type(4)))  int    v4i;

// gfx1250 async global->LDS path (guarded; falls back to manual staging)
#if defined(__gfx1250__) && \
    __has_builtin(__builtin_amdgcn_global_load_async_to_lds_b128) && \
    __has_builtin(__builtin_amdgcn_s_wait_asynccnt)
#define USE_ASYNC_LDS 1
#define AS1 __attribute__((address_space(1)))
#define AS3 __attribute__((address_space(3)))
#else
#define USE_ASYNC_LDS 0
#endif

__device__ __forceinline__ unsigned short f2bf(float f) {
  unsigned u = __float_as_uint(f);
  u += 0x7FFFu + ((u >> 16) & 1u);          // round-to-nearest-even
  return (unsigned short)(u >> 16);
}
__device__ __forceinline__ float siluf(float x) { return x / (1.0f + expf(-x)); }
__device__ __forceinline__ float softplusf(float x) {
  return (x > 20.0f) ? x : log1pf(expf(x));
}

// ---------------- weight / activation precision conversion ----------------
__global__ void k_f32_to_bf16(const float* __restrict__ src,
                              unsigned short* __restrict__ dst, long n) {
  long i = (long)blockIdx.x * blockDim.x + threadIdx.x;
  long stride = (long)gridDim.x * blockDim.x;
  for (; i < n; i += stride) dst[i] = f2bf(src[i]);
}

// ---------------- embedding gather ----------------
__global__ void k_embed(const int* __restrict__ ids, const float* __restrict__ emb,
                        float* __restrict__ x) {
  int t = blockIdx.x;
  long row = ids[t];
  for (int d = threadIdx.x; d < D_MODEL; d += blockDim.x)
    x[(long)t * D_MODEL + d] = emb[row * D_MODEL + d];
}

// ---------------- LUT rmsnorm -> bf16 (blockDim must be 256) ----------------
__global__ __launch_bounds__(256) void k_rmsnorm_lut_bf16(
    const float* __restrict__ x, const float* __restrict__ w,
    unsigned short* __restrict__ out, int dim) {
  __shared__ float red[256];
  int t = blockIdx.x;
  const float* xr = x + (long)t * dim;
  float ss = 0.f;
  for (int i = threadIdx.x; i < dim; i += 256) { float v = xr[i]; ss += v * v; }
  red[threadIdx.x] = ss; __syncthreads();
  for (int o = 128; o > 0; o >>= 1) {
    if (threadIdx.x < o) red[threadIdx.x] += red[threadIdx.x + o];
    __syncthreads();
  }
  float rms = red[0] / (float)dim + EPS;
  const float x0 = 1e-5f, step = (10.0f - 1e-5f) / 255.0f;
  int idx = (int)ceilf((rms - x0) / step);          // searchsorted(left) on linspace
  idx = max(0, min(255, idx));
  float scale = rsqrtf(x0 + (float)idx * step);     // LUT_Y[idx]
  for (int i = threadIdx.x; i < dim; i += 256)
    out[(long)t * dim + i] = f2bf(xr[i] * scale * w[i]);
}

// ---------------- bf16 WMMA GEMM: C[MxN] = A[MxK] * W[NxK]^T (+ residual) ----
// 256 threads = 8 waves; block tile 128x64; wave tile 32x32 (4 WMMA accs).
// LDS rows = 5 uint4 chunks (4 data + 1 pad) = 40 bf16 -> conflict-free b128s.
// Double-buffered tiles; async global->LDS overlapped with WMMA compute.
__global__ __launch_bounds__(256) void k_gemm_bf16(
    const unsigned short* __restrict__ A, const unsigned short* __restrict__ W,
    float* __restrict__ C, const float* __restrict__ residual,
    int M, int N, int K) {
  __shared__ uint4 ldsA[2][128 * 5];
  __shared__ uint4 ldsW[2][64 * 5];
  const int tid  = threadIdx.x;
  const int lane = tid & 31;
  const int wid  = tid >> 5;
  const int wm   = wid & 3;          // 32-row slice within block tile
  const int wn   = wid >> 2;         // 32-col slice within block tile
  const int m0   = blockIdx.y * 128;
  const int n0   = blockIdx.x * 64;

  // staging map: 4 threads per row, one 8xbf16 chunk each
  const int ac    = tid & 3;         // chunk 0..3
  const int arow0 = tid >> 2;        // A rows 0..63
  const int arow1 = arow0 + 64;      // A rows 64..127
  const int wrow  = tid >> 2;        // W rows 0..63
  const int wr    = n0 + wrow;
  const bool wvalid = wr < N;

  const unsigned short* Asrc0 = A + (long)(m0 + arow0) * K;
  const unsigned short* Asrc1 = A + (long)(m0 + arow1) * K;
  // ragged rows: clamp source to row 0 (always valid), dump into the unread
  // pad chunk (index 4) so every wave issues exactly 3 async ops per tile.
  const unsigned short* Wsrc = W + (long)(wvalid ? wr : 0) * K;
  const int wdst = wvalid ? (wrow * 5 + ac) : (wrow * 5 + 4);

  // zero ragged W rows ONCE per buffer; never re-staged inside the K loop
  if (!wvalid) {
    uint4 z = {0u, 0u, 0u, 0u};
    ldsW[0][wrow * 5 + ac] = z;
    ldsW[1][wrow * 5 + ac] = z;
  }

  auto stage = [&](int buf, int k0) {
#if USE_ASYNC_LDS
    __builtin_amdgcn_global_load_async_to_lds_b128(
        (AS1 v4i*)((const uint4*)(Asrc0 + k0) + ac),
        (AS3 v4i*)&ldsA[buf][arow0 * 5 + ac], 0, 0);
    __builtin_amdgcn_global_load_async_to_lds_b128(
        (AS1 v4i*)((const uint4*)(Asrc1 + k0) + ac),
        (AS3 v4i*)&ldsA[buf][arow1 * 5 + ac], 0, 0);
    __builtin_amdgcn_global_load_async_to_lds_b128(
        (AS1 v4i*)((const uint4*)(Wsrc + k0) + ac),
        (AS3 v4i*)&ldsW[buf][wdst], 0, 0);
#else
    ldsA[buf][arow0 * 5 + ac] = ((const uint4*)(Asrc0 + k0))[ac];
    ldsA[buf][arow1 * 5 + ac] = ((const uint4*)(Asrc1 + k0))[ac];
    ldsW[buf][wdst]           = ((const uint4*)(Wsrc + k0))[ac];
#endif
  };

  v8f c00 = {}, c01 = {}, c10 = {}, c11 = {};

  const int nk = K >> 5;             // K / 32 tiles
  stage(0, 0);
  for (int i = 0; i < nk; i++) {
    const int cur = i & 1;
    if (i + 1 < nk) {
      stage(1 - cur, (i + 1) << 5);  // prefetch next tile into other buffer
#if USE_ASYNC_LDS
      __builtin_amdgcn_s_wait_asynccnt(3);   // tile i landed; tile i+1 in flight
#endif
    } else {
#if USE_ASYNC_LDS
      __builtin_amdgcn_s_wait_asynccnt(0);
#endif
    }
    __syncthreads();

    // A frag 16x32: lane<16 -> K 0..7 & 16..23 ; lane>=16 -> K 8..15 & 24..31
    const int la   = lane & 15;
    const int ksel = lane >> 4;
    const int ra   = wm * 32 + la;
    union F { uint4 u[2]; v16bf v; } a0, a1, b0, b1;
    a0.u[0] = ldsA[cur][ra * 5 + ksel];        a0.u[1] = ldsA[cur][ra * 5 + ksel + 2];
    a1.u[0] = ldsA[cur][(ra + 16) * 5 + ksel]; a1.u[1] = ldsA[cur][(ra + 16) * 5 + ksel + 2];
    // B frag 32x16: lane<16 -> K 0..7 & 16..23 ; lane>=16 -> K 8..15 & 24..31 of column n
    const int cb0 = wn * 32 + la;
    const int cb1 = cb0 + 16;
    b0.u[0] = ldsW[cur][cb0 * 5 + ksel]; b0.u[1] = ldsW[cur][cb0 * 5 + ksel + 2];
    b1.u[0] = ldsW[cur][cb1 * 5 + ksel]; b1.u[1] = ldsW[cur][cb1 * 5 + ksel + 2];

    c00 = __builtin_amdgcn_wmma_f32_16x16x32_bf16(false, a0.v, false, b0.v,
                                                  (short)0, c00, false, false);
    c01 = __builtin_amdgcn_wmma_f32_16x16x32_bf16(false, a0.v, false, b1.v,
                                                  (short)0, c01, false, false);
    c10 = __builtin_amdgcn_wmma_f32_16x16x32_bf16(false, a1.v, false, b0.v,
                                                  (short)0, c10, false, false);
    c11 = __builtin_amdgcn_wmma_f32_16x16x32_bf16(false, a1.v, false, b1.v,
                                                  (short)0, c11, false, false);
    __syncthreads();                  // done reading buf[cur]; i+1 may restage it
  }

  // C/D layout: VGPR i -> M = i + 8*(lane>=16), N = lane&15
  const int mb0 = m0 + wm * 32 + (lane >> 4) * 8;  // rows for c0x
  const int mb1 = mb0 + 16;                        // rows for c1x
  const int ncA = n0 + wn * 32 + (lane & 15);
  const int ncB = ncA + 16;
  #pragma unroll
  for (int i = 0; i < 8; i++) {
    int r0 = mb0 + i, r1 = mb1 + i;
    if (ncA < N) {
      float v0 = c00[i], v1 = c10[i];
      if (residual) { v0 += residual[(long)r0 * N + ncA]; v1 += residual[(long)r1 * N + ncA]; }
      C[(long)r0 * N + ncA] = v0;
      C[(long)r1 * N + ncA] = v1;
    }
    if (ncB < N) {
      float v0 = c01[i], v1 = c11[i];
      if (residual) { v0 += residual[(long)r0 * N + ncB]; v1 += residual[(long)r1 * N + ncB]; }
      C[(long)r0 * N + ncB] = v0;
      C[(long)r1 * N + ncB] = v1;
    }
  }
}

// ---------------- dt = softplus(zx[..., -NHEADS:] + dt_bias) ----------------
__global__ void k_dt(const float* __restrict__ zx, const float* __restrict__ dtb,
                     float* __restrict__ dt) {
  int i = blockIdx.x * blockDim.x + threadIdx.x;
  if (i >= SEQ * NHEADS) return;
  int t = i / NHEADS, h = i % NHEADS;
  dt[i] = softplusf(zx[(long)t * D_IN_PROJ + D_INNER + CONV_DIM + h] + dtb[h]);
}

// ---------------- global abs-max over xBC slice (for _quant_act) ------------
__global__ void k_zero(float* p) { if (threadIdx.x == 0 && blockIdx.x == 0) *p = 0.f; }

__global__ __launch_bounds__(256) void k_absmax(const float* __restrict__ zx,
                                                float* __restrict__ amax) {
  __shared__ float red[256];
  const float* row = zx + (long)blockIdx.x * D_IN_PROJ + D_INNER;  // one token/block
  float m = 0.f;
  for (int i = threadIdx.x; i < CONV_DIM; i += 256) m = fmaxf(m, fabsf(row[i]));
  red[threadIdx.x] = m; __syncthreads();
  for (int o = 128; o > 0; o >>= 1) {
    if (threadIdx.x < o) red[threadIdx.x] = fmaxf(red[threadIdx.x], red[threadIdx.x + o]);
    __syncthreads();
  }
  if (threadIdx.x == 0)
    atomicMax((unsigned int*)amax, __float_as_uint(red[0])); // values >= 0
}

// ---------------- fake-quant depthwise conv + SiLU --------------------------
__global__ void k_conv(const float* __restrict__ zx, const float* __restrict__ qcw,
                       const float* __restrict__ cscale_p, const float* __restrict__ cb,
                       const float* __restrict__ amax, float* __restrict__ xbc_out) {
  long i = (long)blockIdx.x * blockDim.x + threadIdx.x;
  if (i >= (long)SEQ * CONV_DIM) return;
  int t = (int)(i / CONV_DIM), c = (int)(i % CONV_DIM);
  float s = amax[0] / (127.0f + 1e-8f);
  float inv_s = 1.0f / s;
  float acc = cb[c];
  #pragma unroll
  for (int k = 0; k < D_CONV; k++) {
    int tt = t + k - (D_CONV - 1);
    if (tt < 0) continue;                   // zero-pad contributes 0
    float v = zx[(long)tt * D_IN_PROJ + D_INNER + c];
    float q = fminf(fmaxf(rintf(v * inv_s), -128.f), 127.f);
    acc += q * qcw[c * D_CONV + k];
  }
  xbc_out[i] = siluf(acc * (cscale_p[0] * s));
}

// ---------------- sequential SSM scan: one block per head -------------------
__global__ __launch_bounds__(256) void k_scan(
    const float* __restrict__ xbc, const float* __restrict__ dt,
    const float* __restrict__ A_log, const float* __restrict__ D_param,
    float* __restrict__ y) {
  __shared__ float sB[D_STATE], sC[D_STATE], sX[HEADDIM], sP[256];
  const int head = blockIdx.x;
  const int tid = threadIdx.x;
  const int p = tid & 63, g = tid >> 6;     // p = headdim index, g = state chunk
  const float negA = -expf(A_log[head]);
  const float Dp = D_param[head];
  float h[32];
  #pragma unroll
  for (int j = 0; j < 32; j++) h[j] = 0.f;

  for (int t = 0; t < SEQ; t++) {
    const float* row = xbc + (long)t * CONV_DIM;
    if (tid < 128)      sB[tid]       = row[D_INNER + tid];
    else                sC[tid - 128] = row[D_INNER + D_STATE + (tid - 128)];
    if (tid < 64)       sX[tid]       = row[head * HEADDIM + tid];
    __syncthreads();
    float dt_t = dt[t * NHEADS + head];
    float edA  = expf(dt_t * negA);
    float xdt  = sX[p] * dt_t;
    float acc  = 0.f;
    #pragma unroll
    for (int j = 0; j < 32; j++) {
      int n = g * 32 + j;
      h[j] = edA * h[j] + xdt * sB[n];
      acc += h[j] * sC[n];
    }
    sP[tid] = acc;
    __syncthreads();
    if (g == 0)
      y[(long)t * D_INNER + head * HEADDIM + p] =
          sP[p] + sP[p + 64] + sP[p + 128] + sP[p + 192] + sX[p] * Dp;
    __syncthreads();
  }
}

// ---------------- y*silu(z), rmsnorm(mixer_norm_w) -> bf16 ------------------
__global__ __launch_bounds__(256) void k_gate_norm(
    const float* __restrict__ y, const float* __restrict__ zx,
    const float* __restrict__ mnw, unsigned short* __restrict__ out) {
  __shared__ float red[256];
  int t = blockIdx.x;
  const float* yr = y  + (long)t * D_INNER;
  const float* zr = zx + (long)t * D_IN_PROJ;   // z = first D_INNER cols
  float vals[6]; float ss = 0.f;
  #pragma unroll
  for (int j = 0; j < 6; j++) {
    int i = threadIdx.x + j * 256;
    float v = yr[i] * siluf(zr[i]);
    vals[j] = v; ss += v * v;
  }
  red[threadIdx.x] = ss; __syncthreads();
  for (int o = 128; o > 0; o >>= 1) {
    if (threadIdx.x < o) red[threadIdx.x] += red[threadIdx.x + o];
    __syncthreads();
  }
  float scale = rsqrtf(red[0] / (float)D_INNER + EPS);
  #pragma unroll
  for (int j = 0; j < 6; j++) {
    int i = threadIdx.x + j * 256;
    out[(long)t * D_INNER + i] = f2bf(vals[j] * scale * mnw[i]);
  }
}

// ============================ host orchestration ============================
extern "C" void kernel_launch(void* const* d_in, const int* in_sizes, int n_in,
                              void* d_out, int out_size, void* d_ws, size_t ws_size,
                              hipStream_t stream) {
  const int*   ids    = (const int*)  d_in[0];
  const float* emb    = (const float*)d_in[1];
  const float* norm_w = (const float*)d_in[2];
  const float* inW    = (const float*)d_in[3];
  const float* qcw    = (const float*)d_in[4];
  const float* cscale = (const float*)d_in[5];
  const float* cb     = (const float*)d_in[6];
  const float* dtb    = (const float*)d_in[7];
  const float* A_log  = (const float*)d_in[8];
  const float* D_par  = (const float*)d_in[9];
  const float* mnw    = (const float*)d_in[10];
  const float* outW   = (const float*)d_in[11];
  const float* normf  = (const float*)d_in[12];
  const float* lmW    = (const float*)d_in[13];
  float* out = (float*)d_out;

  char* base = (char*)d_ws;
  size_t off = 0;
  auto carve = [&](size_t bytes) -> void* {
    void* p = base + off;
    off += (bytes + 255) & ~(size_t)255;
    return p;
  };
  unsigned short* wIn  = (unsigned short*)carve((size_t)N_LAYER * D_IN_PROJ * D_MODEL * 2);
  unsigned short* wOut = (unsigned short*)carve((size_t)N_LAYER * D_MODEL * D_INNER * 2);
  unsigned short* wLm  = (unsigned short*)carve((size_t)VOCAB * D_MODEL * 2);
  float*          x    = (float*)         carve((size_t)SEQ * D_MODEL * 4);
  unsigned short* uBf  = (unsigned short*)carve((size_t)SEQ * D_MODEL * 2);
  float*          zx   = (float*)         carve((size_t)SEQ * D_IN_PROJ * 4);
  float*          xbc  = (float*)         carve((size_t)SEQ * CONV_DIM * 4);
  float*          dtv  = (float*)         carve((size_t)SEQ * NHEADS * 4);
  float*          yv   = (float*)         carve((size_t)SEQ * D_INNER * 4);
  unsigned short* yBf  = (unsigned short*)carve((size_t)SEQ * D_INNER * 2);
  float*          amax = (float*)         carve(256);
  (void)in_sizes; (void)n_in; (void)out_size; (void)ws_size;

  // bf16 weight images (rebuilt every call; deterministic)
  k_f32_to_bf16<<<4096, 256, 0, stream>>>(inW,  wIn,  (long)N_LAYER * D_IN_PROJ * D_MODEL);
  k_f32_to_bf16<<<4096, 256, 0, stream>>>(outW, wOut, (long)N_LAYER * D_MODEL * D_INNER);
  k_f32_to_bf16<<<8192, 256, 0, stream>>>(lmW,  wLm,  (long)VOCAB * D_MODEL);

  k_embed<<<SEQ, 256, 0, stream>>>(ids, emb, x);

  dim3 g_in ((D_IN_PROJ + 63) / 64, SEQ / 128);
  dim3 g_out((D_MODEL   + 63) / 64, SEQ / 128);
  dim3 g_lm ((VOCAB     + 63) / 64, SEQ / 128);

  for (int i = 0; i < N_LAYER; i++) {
    k_rmsnorm_lut_bf16<<<SEQ, 256, 0, stream>>>(x, norm_w + (size_t)i * D_MODEL, uBf, D_MODEL);
    k_gemm_bf16<<<g_in, 256, 0, stream>>>(uBf, wIn + (size_t)i * D_IN_PROJ * D_MODEL,
                                          zx, nullptr, SEQ, D_IN_PROJ, D_MODEL);
    k_dt<<<(SEQ * NHEADS + 255) / 256, 256, 0, stream>>>(zx, dtb + (size_t)i * NHEADS, dtv);
    k_zero<<<1, 1, 0, stream>>>(amax);
    k_absmax<<<SEQ, 256, 0, stream>>>(zx, amax);
    k_conv<<<(SEQ * CONV_DIM + 255) / 256, 256, 0, stream>>>(
        zx, qcw + (size_t)i * CONV_DIM * D_CONV, cscale + i,
        cb + (size_t)i * CONV_DIM, amax, xbc);
    k_scan<<<NHEADS, 256, 0, stream>>>(xbc, dtv, A_log + (size_t)i * NHEADS,
                                       D_par + (size_t)i * NHEADS, yv);
    k_gate_norm<<<SEQ, 256, 0, stream>>>(yv, zx, mnw + (size_t)i * D_INNER, yBf);
    // x = x + out_proj(y)   (in-place residual: each element touched by one thread)
    k_gemm_bf16<<<g_out, 256, 0, stream>>>(yBf, wOut + (size_t)i * D_MODEL * D_INNER,
                                           x, x, SEQ, D_MODEL, D_INNER);
  }

  k_rmsnorm_lut_bf16<<<SEQ, 256, 0, stream>>>(x, normf, uBf, D_MODEL);
  k_gemm_bf16<<<g_lm, 256, 0, stream>>>(uBf, wLm, out, nullptr, SEQ, VOCAB, D_MODEL);
}
